// ShiftedWindowAttention_16793322127482
// MI455X (gfx1250) — matrix operand order repaired and checked by reading, not compiled, not run on hardware
//
#include <hip/hip_runtime.h>
#include <hip/hip_bf16.h>

typedef __attribute__((ext_vector_type(16))) _Float16 v16h;
typedef __attribute__((ext_vector_type(8)))  _Float16 v8h;
typedef __attribute__((ext_vector_type(4)))  _Float16 v4h;
typedef __attribute__((ext_vector_type(8)))  float    v8f;
typedef __attribute__((ext_vector_type(4)))  float    f32x4;

#define BATCH  32
#define HW     56
#define CHN    384
#define NHEAD  12
#define HDIM   32
#define WSZ    7
#define NTOK   49
#define NPAD   64
#define SHIFT  3
#define SCALE_Q 0.17677669529663689f   // 1/sqrt(32)

static __device__ __forceinline__ v8f wmma16(const v16h& a, const v16h& b, const v8f& c) {
  // D = A(16x32 f16) * B(32x16 f16) + C(16x16 f32)
  return __builtin_amdgcn_wmma_f32_16x16x32_f16(false, a, false, b, (short)0, c, false, false);
}

static __device__ __forceinline__ v16h cat8(const v8h& lo, const v8h& hi) {
  return __builtin_shufflevector(lo, hi, 0, 1, 2, 3, 4, 5, 6, 7,
                                 8, 9, 10, 11, 12, 13, 14, 15);
}

// A fragment 16x32 f16 from row-major [M][K] storage (ld = K stride, multiples of 8).
// lane&15 = row; lane>>4 selects K 0-15 vs 16-31 (contiguous 32 B per lane).
static __device__ __forceinline__ v16h loadA(const _Float16* base, int ld, int lane,
                                             int mbase, int kbase) {
  int m = mbase + (lane & 15);
  int g = lane >> 4;
  const v8h* p = (const v8h*)(base + (size_t)m * ld + kbase + g * 16);
  return cat8(p[0], p[1]);   // K = kb+g*16..+15
}

// B fragment 32x16 (KxN) where memory is row-major W[n][k] => B[k][n] = W[n][k].
// lane&15 = column n; lane group selects K 0-15 vs 16-31 (contiguous 32 B per lane).
static __device__ __forceinline__ v16h loadB_rows(const _Float16* base, int ld, int lane,
                                                  int nbase, int kbase) {
  int n = nbase + (lane & 15);
  const v8h* p = (const v8h*)(base + (size_t)n * ld + kbase + (lane >> 4) * 16);
  return cat8(p[0], p[1]);
}

// C/D layout: VGPR r, lane l -> M = mbase + r + 8*(l>>4), N = nbase + (l&15)
static __device__ __forceinline__ void storeC_f16_bias(_Float16* base, int ld, int lane,
                                                       int mbase, int nbase, const v8f& c,
                                                       float addv) {
  int n = nbase + (lane & 15);
  int g = lane >> 4;
#pragma unroll
  for (int r = 0; r < 8; ++r)
    base[(size_t)(mbase + r + 8 * g) * ld + n] = (_Float16)(c[r] + addv);
}

static __device__ __forceinline__ void storeC_f16(_Float16* base, int ld, int lane,
                                                  int mbase, int nbase, const v8f& c) {
  int n = nbase + (lane & 15);
  int g = lane >> 4;
#pragma unroll
  for (int r = 0; r < 8; ++r)
    base[(size_t)(mbase + r + 8 * g) * ld + n] = (_Float16)c[r];
}

// Transposed C store: baseT is [N][M] with row stride ld. Per lane the 8 M-values are
// consecutive -> single 16-byte ds_store_b128.
static __device__ __forceinline__ void storeC_f16_T(_Float16* baseT, int ld, int lane,
                                                    int mbase, int nbase, const v8f& c,
                                                    float addv) {
  int n = nbase + (lane & 15);
  int g = lane >> 4;
  v8h h;
#pragma unroll
  for (int r = 0; r < 8; ++r) h[r] = (_Float16)(c[r] + addv);
  *(v8h*)(baseT + (size_t)n * ld + mbase + 8 * g) = h;
}

static __device__ __forceinline__ void storeC_f32(float* base, int ld, int lane,
                                                  int mbase, int nbase, const v8f& c) {
  int n = nbase + (lane & 15);
  int g = lane >> 4;
#pragma unroll
  for (int r = 0; r < 8; ++r)
    base[(size_t)(mbase + r + 8 * g) * ld + n] = c[r];
}

static __device__ __forceinline__ int region_of(int R, int Cc) {
  int hr = (R < HW - WSZ) ? 0 : ((R < HW - SHIFT) ? 1 : 2);
  int wr = (Cc < HW - WSZ) ? 0 : ((Cc < HW - SHIFT) ? 1 : 2);
  return hr * 3 + wr;
}

// ---------------- prep: f32 weights -> f16, gather relative-position bias -------------
__global__ __launch_bounds__(256) void swin_prep_kernel(
    const float* __restrict__ wqkv_f, const float* __restrict__ wproj_f,
    const float* __restrict__ btab, const int* __restrict__ ridx,
    _Float16* __restrict__ wqkv_h, _Float16* __restrict__ wproj_h,
    float* __restrict__ biasT) {
  int i = blockIdx.x * blockDim.x + threadIdx.x;
  int stride = gridDim.x * blockDim.x;
  for (int t = i; t < 3 * CHN * CHN; t += stride) wqkv_h[t] = (_Float16)wqkv_f[t];
  for (int t = i; t < CHN * CHN; t += stride) wproj_h[t] = (_Float16)wproj_f[t];
  for (int t = i; t < NHEAD * NTOK * NTOK; t += stride) {
    int h = t / (NTOK * NTOK);
    int p = t - h * NTOK * NTOK;
    biasT[t] = btab[(size_t)ridx[p] * NHEAD + h];
  }
}

// ---------------- fused shifted-window attention: one workgroup per window -----------
__global__ __launch_bounds__(256) void swin_attn_kernel(
    const float* __restrict__ x, const _Float16* __restrict__ wqkv,
    const float* __restrict__ qkvb, const _Float16* __restrict__ wproj,
    const float* __restrict__ projb, const float* __restrict__ biasT,
    float* __restrict__ out) {
  __shared__ __align__(16) _Float16 xs[NPAD * CHN];    // 48 KB shifted window, f16
  __shared__ __align__(16) _Float16 qs[NPAD * HDIM];   //  4 KB q  [token][d]
  __shared__ __align__(16) _Float16 ks[NPAD * HDIM];   //  4 KB k  [token][d]
  __shared__ __align__(16) _Float16 vsT[HDIM * NPAD];  //  4 KB vT [d][token]
  __shared__ __align__(16) float    Sb[NPAD * NPAD];   // 16 KB logits
  __shared__ __align__(16) _Float16 Pb[NPAD * NPAD];   //  8 KB probs, f16
  __shared__ __align__(16) _Float16 os[NPAD * CHN];    // 48 KB attention output
  __shared__ int regid[NPAD];                          // shift-mask region per token

  const int w    = blockIdx.x;
  const int b    = w >> 6;          // batch
  const int widx = w & 63;
  const int wh   = widx >> 3;
  const int ww   = widx & 7;
  const int tid  = threadIdx.x;
  const int lane = tid & 31;
  const int wv   = tid >> 5;        // 8 waves

  __builtin_prefetch(wqkv, 0, 1);
  __builtin_prefetch(wproj, 0, 1);

  // region id table for the shift mask (only edge windows mix regions)
  if (tid < NPAD) {
    int t = tid;
    int r = t / WSZ, c = t - r * WSZ;
    regid[tid] = (t < NTOK) ? region_of(wh * WSZ + r, ww * WSZ + c) : 0;
  }

  // ---- load window with cyclic shift (roll -3), f32 -> f16, 4-wide ----
  for (int idx = tid; idx < NTOK * (CHN / 4); idx += 256) {
    int t  = idx / (CHN / 4);
    int c4 = idx - t * (CHN / 4);
    int r  = t / WSZ, c = t - r * WSZ;
    int gr = (wh * WSZ + r + SHIFT) % HW;
    int gc = (ww * WSZ + c + SHIFT) % HW;
    f32x4 vv = *(const f32x4*)&x[(((size_t)b * HW + gr) * HW + gc) * CHN + c4 * 4];
    v4h hh;
#pragma unroll
    for (int k = 0; k < 4; ++k) hh[k] = (_Float16)vv[k];
    *(v4h*)&xs[(size_t)t * CHN + c4 * 4] = hh;
  }
  for (int idx = tid; idx < (NPAD - NTOK) * (CHN / 8); idx += 256) {
    v8h z = {};
    *(v8h*)&xs[NTOK * CHN + idx * 8] = z;
  }
  __syncthreads();

  // ---- per-head: QKV projection -> S = qk^T -> softmax -> O = Pv ----
  for (int h = 0; h < NHEAD; ++h) {
    // QKV: 3 (q,k,v) x 4 M-tiles x 2 N-tiles = 24 tiles; 3 per wave
#pragma unroll
    for (int t3 = 0; t3 < 3; ++t3) {
      int id = wv * 3 + t3;
      int s = id >> 3, rem = id & 7, mt = rem >> 1, nt = rem & 1;
      const _Float16* Wbase = wqkv + (size_t)(s * CHN + h * HDIM + nt * 16) * CHN;
      v8f acc = {};
      for (int kb = 0; kb < CHN; kb += 32) {
        v16h a  = loadA(xs, CHN, lane, mt * 16, kb);
        v16h bf = loadB_rows(Wbase, CHN, lane, 0, kb);
        acc = wmma16(a, bf, acc);
      }
      float bias = qkvb[s * CHN + h * HDIM + nt * 16 + (lane & 15)];
      if (s == 0)      storeC_f16_bias(qs,  HDIM, lane, mt * 16, nt * 16, acc, bias);
      else if (s == 1) storeC_f16_bias(ks,  HDIM, lane, mt * 16, nt * 16, acc, bias);
      else             storeC_f16_T  (vsT, NPAD, lane, mt * 16, nt * 16, acc, bias);
    }
    __syncthreads();

    // S = q @ k^T : 4x4 tiles, K = 32 (one WMMA); 2 tiles per wave
#pragma unroll
    for (int t2 = 0; t2 < 2; ++t2) {
      int id = wv * 2 + t2;
      int mt = id >> 2, nt = id & 3;
      v16h a  = loadA(qs, HDIM, lane, mt * 16, 0);
      v16h bf = loadB_rows(ks, HDIM, lane, nt * 16, 0);  // B[k][n] = k[n][k]
      v8f acc = {};
      acc = wmma16(a, bf, acc);
      storeC_f32(Sb, NPAD, lane, mt * 16, nt * 16, acc);
    }
    __syncthreads();

    // softmax per row; skewed column order (j = (jj+i)&63) -> conflict-free LDS banks.
    // Pass 1 folds scale + rel-pos bias + shift mask into Sb in place while tracking max.
    if (tid < NPAD) {
      int i = tid;
      if (i < NTOK) {
        int idi = regid[i];
        const float* brow = biasT + (size_t)h * (NTOK * NTOK) + (size_t)i * NTOK;
        float mx = -1e30f;
        for (int jj = 0; jj < NPAD; ++jj) {
          int j = (jj + i) & 63;
          if (j < NTOK) {
            float m = (idi != regid[j]) ? -100.f : 0.f;
            float sv = Sb[i * NPAD + j] * SCALE_Q + brow[j] + m;
            Sb[i * NPAD + j] = sv;
            mx = fmaxf(mx, sv);
          }
        }
        float sum = 0.f;
        for (int jj = 0; jj < NPAD; ++jj) {
          int j = (jj + i) & 63;
          if (j < NTOK) {
            float e = __expf(Sb[i * NPAD + j] - mx);
            sum += e;
            Pb[i * NPAD + j] = (_Float16)e;
          } else {
            Pb[i * NPAD + j] = (_Float16)0.f;
          }
        }
        float inv = 1.f / sum;
        for (int jj = 0; jj < NPAD; ++jj) {
          int j = (jj + i) & 63;
          if (j < NTOK)
            Pb[i * NPAD + j] = (_Float16)((float)Pb[i * NPAD + j] * inv);
        }
      } else {
        for (int jj = 0; jj < NPAD; ++jj) {
          int j = (jj + i) & 63;
          Pb[i * NPAD + j] = (_Float16)0.f;
        }
      }
    }
    __syncthreads();

    // O = P @ v : 4x2 tiles, K = 64 (two WMMA); 1 tile per wave.
    // vT[d][token] is row-major W[n][k] for B[k][n]=v[k][n] -> contiguous B loads.
    {
      int mt = wv >> 1, nt = wv & 1;
      v8f acc = {};
#pragma unroll
      for (int kb = 0; kb < NPAD; kb += 32) {
        v16h a  = loadA(Pb, NPAD, lane, mt * 16, kb);
        v16h bf = loadB_rows(vsT, NPAD, lane, nt * 16, kb);
        acc = wmma16(a, bf, acc);
      }
      storeC_f16(os + h * HDIM, CHN, lane, mt * 16, nt * 16, acc);
    }
    __syncthreads();
  }

  // ---- output projection + window reverse (roll +3) ----
  for (int t12 = 0; t12 < 12; ++t12) {
    int id = wv * 12 + t12;
    int mt = id / 24, nt = id % 24;
    const _Float16* Wbase = wproj + (size_t)(nt * 16) * CHN;
    v8f acc = {};
    for (int kb = 0; kb < CHN; kb += 32) {
      v16h a  = loadA(os, CHN, lane, mt * 16, kb);
      v16h bf = loadB_rows(Wbase, CHN, lane, 0, kb);
      acc = wmma16(a, bf, acc);
    }
    int n = nt * 16 + (lane & 15);
    float pb = projb[n];
    int g = lane >> 4;
#pragma unroll
    for (int r8 = 0; r8 < 8; ++r8) {
      int m = mt * 16 + r8 + 8 * g;
      if (m < NTOK) {
        int rr = m / WSZ, cc = m % WSZ;
        int gr = (wh * WSZ + rr + SHIFT) % HW;
        int gc = (ww * WSZ + cc + SHIFT) % HW;
        out[(((size_t)b * HW + gr) * HW + gc) * CHN + n] = acc[r8] + pb;
      }
    }
  }
}

extern "C" void kernel_launch(void* const* d_in, const int* in_sizes, int n_in,
                              void* d_out, int out_size, void* d_ws, size_t ws_size,
                              hipStream_t stream) {
  const float* x     = (const float*)d_in[0];
  const float* wqkvf = (const float*)d_in[1];
  const float* qkvb  = (const float*)d_in[2];
  const float* wprjf = (const float*)d_in[3];
  const float* prjb  = (const float*)d_in[4];
  const float* btab  = (const float*)d_in[5];
  const int*   ridx  = (const int*)d_in[6];
  float* out = (float*)d_out;

  // workspace layout (bytes): f16 qkv weight | f16 proj weight | f32 bias [12][49][49]
  char* ws = (char*)d_ws;
  _Float16* wqkv_h = (_Float16*)ws;                               // 884736 B
  _Float16* wprj_h = (_Float16*)(ws + 884736);                    // 294912 B
  float*    biasT  = (float*)(ws + 884736 + 294912);              // 115248 B

  swin_prep_kernel<<<512, 256, 0, stream>>>(wqkvf, wprjf, btab, ridx,
                                            wqkv_h, wprj_h, biasT);

  const int nblocks = BATCH * 64;  // one workgroup per (batch, window)
  swin_attn_kernel<<<nblocks, 256, 0, stream>>>(x, wqkv_h, qkvb, wprj_h,
                                                prjb, biasT, out);
}
